// RFMNet_43319040147715
// MI455X (gfx1250) — matrix operand order repaired and checked by reading, not compiled, not run on hardware
//
#include <hip/hip_runtime.h>

// ---------------------------------------------------------------------------
// RFMNet (DiffusionNet-style) forward for MI455X / gfx1250.
// All matmuls via v_wmma_f32_16x16x32_bf16 (wave32 WMMA, f32 accumulate),
// f32->bf16 packed conversion (cvt_pk / v_perm), LDS double-buffered tiles.
// ---------------------------------------------------------------------------

typedef __attribute__((ext_vector_type(16))) __bf16        v16bf;
typedef __attribute__((ext_vector_type(2)))  __bf16        v2bf;
typedef __attribute__((ext_vector_type(8)))  float         v8f;
typedef __attribute__((ext_vector_type(8)))  unsigned int  v8u;
typedef __attribute__((ext_vector_type(4)))  unsigned int  v4u;

#define NB      4       // N_BLOCK
#define BATCH   4       // B
#define NV      4096    // V
#define NK      128     // K (spectral)
#define CW      128     // C_W == C_OUT == N of every GEMM
#define CIN     16
#define ROWS    (BATCH * NV)   // 16384

// f32 -> bf16 (RNE), scalar software fallback
__device__ __forceinline__ unsigned short f2bf(float f) {
  unsigned int u = __builtin_bit_cast(unsigned int, f);
  u += 0x7FFFu + ((u >> 16) & 1u);
  return (unsigned short)(u >> 16);
}

// two f32 -> packed 2xbf16 in one dword.
// Preference: native v_cvt_pk_bf16_f32 > round+v_perm_b32 (3 VALU) > software.
__device__ __forceinline__ unsigned int pack2bf(float x, float y) {
#if __has_builtin(__builtin_amdgcn_cvt_pk_bf16_f32)
  v2bf v = __builtin_amdgcn_cvt_pk_bf16_f32(x, y);
  return __builtin_bit_cast(unsigned int, v);
#elif __has_builtin(__builtin_amdgcn_perm)
  // round-to-nearest on the truncated 16 bits, then pick the two high halves
  const unsigned int ux = __builtin_bit_cast(unsigned int, x) + 0x8000u;
  const unsigned int uy = __builtin_bit_cast(unsigned int, y) + 0x8000u;
  return __builtin_amdgcn_perm(uy, ux, 0x07060302u);  // {uy.hi16, ux.hi16}
#else
  return (unsigned)f2bf(x) | ((unsigned)f2bf(y) << 16);
#endif
}

__device__ __forceinline__ v16bf pack16(v4u a, v4u b) {
  v8u u;
  u[0] = a[0]; u[1] = a[1]; u[2] = a[2]; u[3] = a[3];
  u[4] = b[0]; u[5] = b[1]; u[6] = b[2]; u[7] = b[3];
  return __builtin_bit_cast(v16bf, u);
}

// ---------------------------------------------------------------------------
// GEMM: Cout[M x 128] = epi(alpha * A(M x Kd) @ B(Kd x 128))
//   TRANSA: A element [m][k] = G[k*ldA + m] (for evecs^T); never guarded
//   GUARD : K-dim bounds checks (only when Kd % 32 != 0; featurize K=16)
//   M must be a multiple of 128 (true for every launch here).
//   Block: 256 threads / 8 waves; block tile 128x128; wave tile 32x64 (2x4
//   WMMA accumulators). BK=32, LDS double-buffered, register-staged loads.
// ---------------------------------------------------------------------------
template <bool TRANSA, bool GUARD>
__global__ __launch_bounds__(256, 2)
void gemm_wmma_bf16(const float* __restrict__ A, long long strideA, int ldA,
                    const float* __restrict__ Bm, long long strideB,
                    const float* __restrict__ bias,
                    const float* __restrict__ Cin, long long strideCin,
                    float* __restrict__ Cout, long long strideC,
                    int Kd, float alpha, int doRelu) {
  __shared__ __attribute__((aligned(16))) unsigned short ldsA[2][128][40];
  __shared__ __attribute__((aligned(16))) unsigned short ldsB[2][128][40];

  const int tid  = threadIdx.x;
  const int lane = tid & 31;
  const int wave = tid >> 5;
  const int wm   = wave & 3;   // 4 wave-rows of 32
  const int wn   = wave >> 2;  // 2 wave-cols of 64
  const int m0   = blockIdx.x * 128;

  const long long bz = blockIdx.y;
  const float* Ab = A  + bz * strideA;
  const float* Bb = Bm + bz * strideB;
  float*       Cb = Cout + bz * strideC;
  const float* Rb = Cin ? (Cin + bz * strideCin) : nullptr;

  float4 ar[4];                 // non-trans A: 4 consecutive k per row
  float2 ta0[4], ta1[4];        // transA: 2x2 (k x m) micro-blocks
  float2 tb0[4], tb1[4];        // B:      2x2 (k x n) micro-blocks

  auto loadRegs = [&](int k0) {
#pragma unroll
    for (int r = 0; r < 4; ++r) {
      const int i = tid + r * 256;
      if (TRANSA) {
        const int k2 = (i >> 6) << 1, mm2 = (i & 63) * 2;
        ta0[r] = *(const float2*)&Ab[(long long)(k0 + k2 + 0) * ldA + m0 + mm2];
        ta1[r] = *(const float2*)&Ab[(long long)(k0 + k2 + 1) * ldA + m0 + mm2];
      } else {
        const int mm = i >> 3, kk = (i & 7) * 4;        // contiguous in k
        if (GUARD) {
          const long long rowb = (long long)(m0 + mm) * ldA + k0 + kk;
          float x0 = 0.f, x1 = 0.f, x2 = 0.f, x3 = 0.f;
          if (k0 + kk + 0 < Kd) x0 = Ab[rowb + 0];
          if (k0 + kk + 1 < Kd) x1 = Ab[rowb + 1];
          if (k0 + kk + 2 < Kd) x2 = Ab[rowb + 2];
          if (k0 + kk + 3 < Kd) x3 = Ab[rowb + 3];
          ar[r] = make_float4(x0, x1, x2, x3);
        } else {
          ar[r] = *(const float4*)&Ab[(long long)(m0 + mm) * ldA + kk + k0];
        }
      }
      const int k2 = (i >> 6) << 1, nn2 = (i & 63) * 2;
      if (GUARD && (k0 + k2 >= Kd)) {
        tb0[r] = make_float2(0.f, 0.f);
      } else {
        tb0[r] = *(const float2*)&Bb[(long long)(k0 + k2 + 0) * 128 + nn2];
      }
      if (GUARD && (k0 + k2 + 1 >= Kd)) {
        tb1[r] = make_float2(0.f, 0.f);
      } else {
        tb1[r] = *(const float2*)&Bb[(long long)(k0 + k2 + 1) * 128 + nn2];
      }
    }
  };

  auto storeLds = [&](int p) {
#pragma unroll
    for (int r = 0; r < 4; ++r) {
      const int i = tid + r * 256;
      if (TRANSA) {
        const int k2 = (i >> 6) << 1, mm2 = (i & 63) * 2;
        *(unsigned int*)&ldsA[p][mm2 + 0][k2] = pack2bf(ta0[r].x, ta1[r].x);
        *(unsigned int*)&ldsA[p][mm2 + 1][k2] = pack2bf(ta0[r].y, ta1[r].y);
      } else {
        const int mm = i >> 3, kk = (i & 7) * 4;
        *(uint2*)&ldsA[p][mm][kk] =
            make_uint2(pack2bf(ar[r].x, ar[r].y), pack2bf(ar[r].z, ar[r].w));
      }
      const int k2 = (i >> 6) << 1, nn2 = (i & 63) * 2;
      *(unsigned int*)&ldsB[p][nn2 + 0][k2] = pack2bf(tb0[r].x, tb1[r].x);
      *(unsigned int*)&ldsB[p][nn2 + 1][k2] = pack2bf(tb0[r].y, tb1[r].y);
    }
  };

  v8f acc[2][4] = {};
  const int nk = (Kd + 31) >> 5;

  loadRegs(0);
  storeLds(0);
  int p = 0;

  const int lh = lane >> 4;      // half-wave select
  const int lm = lane & 15;
  const int ac0 = lh * 8;        // A: K chunks {0-7,16-23} / {8-15,24-31}
  const int bc0 = lh * 8;        // B: K chunks {0-7,16-23} / {8-15,24-31}

  for (int kt = 0; kt < nk; ++kt) {
    __syncthreads();
    const bool more = (kt + 1 < nk);
    if (more) loadRegs((kt + 1) << 5);
    if (!GUARD) {                              // GL2 prefetch two tiles ahead
      const int kpf = (kt + 2) << 5;
      if (kpf < Kd) {
        if (TRANSA)
          __builtin_prefetch(&Ab[(long long)(kpf + (tid >> 5)) * ldA + m0 + (tid & 31) * 4], 0, 1);
        else
          __builtin_prefetch(&Ab[(long long)(m0 + (tid >> 1)) * ldA + kpf], 0, 1);
      }
    }

    v16bf afr[2], bfr[4];
#pragma unroll
    for (int mt = 0; mt < 2; ++mt) {
      const unsigned short* r = &ldsA[p][wm * 32 + mt * 16 + lm][0];
      afr[mt] = pack16(*(const v4u*)(r + ac0), *(const v4u*)(r + ac0 + 16));
    }
#pragma unroll
    for (int nt = 0; nt < 4; ++nt) {
      const unsigned short* r = &ldsB[p][wn * 64 + nt * 16 + lm][0];
      bfr[nt] = pack16(*(const v4u*)(r + bc0), *(const v4u*)(r + bc0 + 16));
    }
#pragma unroll
    for (int mt = 0; mt < 2; ++mt)
#pragma unroll
      for (int nt = 0; nt < 4; ++nt)
        acc[mt][nt] = __builtin_amdgcn_wmma_f32_16x16x32_bf16(
            false, afr[mt], false, bfr[nt], (short)0, acc[mt][nt], false, false);

    if (more) storeLds(p ^ 1);
    p ^= 1;
  }

  // -- epilogue (M multiple of 128 -> no row guard) --
  const int coll   = lane & 15;
  const int rowsel = (lane >> 4) * 8;
#pragma unroll
  for (int mt = 0; mt < 2; ++mt) {
#pragma unroll
    for (int nt = 0; nt < 4; ++nt) {
      const int col = wn * 64 + nt * 16 + coll;
#pragma unroll
      for (int j = 0; j < 8; ++j) {
        const int rowg = m0 + wm * 32 + mt * 16 + rowsel + j;
        float r = alpha * acc[mt][nt][j];
        if (bias) r += bias[col];
        if (Rb)   r += Rb[(long long)rowg * 128 + col];
        if (doRelu) r = fmaxf(r, 0.f);
        Cb[(long long)rowg * 128 + col] = r;
      }
    }
  }
}

// ---------------------------------------------------------------------------
// Elementwise helpers
// ---------------------------------------------------------------------------
__global__ void mul_mass_kernel(const float* __restrict__ x,
                                const float* __restrict__ mass,
                                float* __restrict__ xm, int n) {
  const int i = blockIdx.x * blockDim.x + threadIdx.x;
  if (i < n) xm[i] = x[i] * mass[i >> 7];
}

// xspec[b,k,c] *= exp(-evals[b,k] * max(t[c],1e-8))
__global__ void spec_scale_kernel(float* __restrict__ xs,
                                  const float* __restrict__ evals,
                                  const float* __restrict__ ti, int n) {
  const int i = blockIdx.x * blockDim.x + threadIdx.x;
  if (i >= n) return;
  const int c = i & 127;
  const int k = (i >> 7) & 127;
  const int b = i >> 14;
  const float tt = fmaxf(ti[c], 1e-8f);
  xs[i] *= __expf(-evals[b * 128 + k] * tt);
}

// f = [x | x_diff | tanh(gx*br + gy*bi)]
__global__ void gfeat_concat_kernel(const float* __restrict__ x,
                                    const float* __restrict__ xd,
                                    const float* __restrict__ gx,
                                    const float* __restrict__ gy,
                                    const float* __restrict__ br,
                                    const float* __restrict__ bi,
                                    float* __restrict__ f, int n) {
  const int i = blockIdx.x * blockDim.x + threadIdx.x;
  if (i >= n) return;
  const int row = i >> 7, c = i & 127;
  const long long base = (long long)row * 384;
  f[base + c]       = x[i];
  f[base + 128 + c] = xd[i];
  f[base + 256 + c] = tanhf(gx[i] * br[i] + gy[i] * bi[i]);
}

// ---------------------------------------------------------------------------
// Host orchestration (graph-capture safe: only kernel launches on `stream`)
// Workspace need: ~84 MB of f32 scratch.
// ---------------------------------------------------------------------------
extern "C" void kernel_launch(void* const* d_in, const int* in_sizes, int n_in,
                              void* d_out, int out_size, void* d_ws, size_t ws_size,
                              hipStream_t stream) {
  (void)in_sizes; (void)n_in; (void)out_size; (void)ws_size;

  const float* x_in  = (const float*)d_in[0];
  const float* mass  = (const float*)d_in[1];
  const float* evals = (const float*)d_in[2];
  const float* evecs = (const float*)d_in[3];
  const float* gradX = (const float*)d_in[4];
  const float* gradY = (const float*)d_in[5];
  const float* Wf    = (const float*)d_in[6];
  const float* bf_   = (const float*)d_in[7];
  const float* Wl    = (const float*)d_in[8];
  const float* bl    = (const float*)d_in[9];
  const float* t     = (const float*)d_in[10];
  const float* Are   = (const float*)d_in[11];
  const float* Aim   = (const float*)d_in[12];
  const float* W0    = (const float*)d_in[13];
  const float* b0    = (const float*)d_in[14];
  const float* W1    = (const float*)d_in[15];
  const float* b1    = (const float*)d_in[16];
  const float* W2    = (const float*)d_in[17];
  const float* b2    = (const float*)d_in[18];

  const size_t RC = (size_t)ROWS * CW;          // 2,097,152 floats
  float* ws    = (float*)d_ws;
  float* xb    = ws;                            // current features x
  float* xm    = xb    + RC;                    // x*mass; reused as h0
  float* xspec = xm    + RC;                    // [B,K,C] 65536 floats
  float* xdiff = xspec + (size_t)BATCH * NK * CW;
  float* gx    = xdiff + RC;                    // reused as h1
  float* gy    = gx    + RC;
  float* br    = gy    + RC;
  float* bi    = br    + RC;
  float* fc    = bi    + RC;                    // [ROWS, 384]

  auto gemm = [&](const float* A, long long sA, int ldA, int tA,
                  const float* B, long long sB,
                  const float* bias, const float* Cin, long long sCin,
                  float* C, long long sC, int M, int Kd, int batch,
                  float alpha, int relu) {
    dim3 grid((unsigned)(M / 128), (unsigned)batch);
    if (tA)
      gemm_wmma_bf16<true, false><<<grid, dim3(256), 0, stream>>>(
          A, sA, ldA, B, sB, bias, Cin, sCin, C, sC, Kd, alpha, relu);
    else if (Kd % 32 != 0)
      gemm_wmma_bf16<false, true><<<grid, dim3(256), 0, stream>>>(
          A, sA, ldA, B, sB, bias, Cin, sCin, C, sC, Kd, alpha, relu);
    else
      gemm_wmma_bf16<false, false><<<grid, dim3(256), 0, stream>>>(
          A, sA, ldA, B, sB, bias, Cin, sCin, C, sC, Kd, alpha, relu);
  };

  const int nel   = ROWS * CW;
  const int nspec = BATCH * NK * CW;

  // featurize: x = x_in @ Wf + bf
  gemm(x_in, 0, CIN, 0, Wf, 0, bf_, nullptr, 0, xb, 0, ROWS, CIN, 1, 1.f, 0);

  for (int i = 0; i < NB; ++i) {
    const float* Are_i = Are + (size_t)i * CW * CW;
    const float* Aim_i = Aim + (size_t)i * CW * CW;

    // xm = x * mass
    mul_mass_kernel<<<nel / 256, 256, 0, stream>>>(xb, mass, xm, nel);
    // x_spec = evecs^T @ xm           (per-batch, M=K=128, Kdim=V)
    gemm(evecs, (long long)NV * NK, NK, 1, xm, (long long)NV * CW,
         nullptr, nullptr, 0, xspec, (long long)NK * CW, NK, NV, BATCH, 1.f, 0);
    // x_spec *= exp(-evals * t_i)
    spec_scale_kernel<<<nspec / 256, 256, 0, stream>>>(xspec, evals, t + i * CW, nspec);
    // x_diff = evecs @ x_spec
    gemm(evecs, (long long)NV * NK, NK, 0, xspec, (long long)NK * CW,
         nullptr, nullptr, 0, xdiff, (long long)NV * CW, NV, NK, BATCH, 1.f, 0);
    // gx = gradX @ x_diff ; gy = gradY @ x_diff   (the big ones)
    gemm(gradX, (long long)NV * NV, NV, 0, xdiff, (long long)NV * CW,
         nullptr, nullptr, 0, gx, (long long)NV * CW, NV, NV, BATCH, 1.f, 0);
    gemm(gradY, (long long)NV * NV, NV, 0, xdiff, (long long)NV * CW,
         nullptr, nullptr, 0, gy, (long long)NV * CW, NV, NV, BATCH, 1.f, 0);
    // complex linear: br = gx@Are - gy@Aim ; bi = gy@Are + gx@Aim
    gemm(gx, 0, CW, 0, Are_i, 0, nullptr, nullptr, 0, br, 0, ROWS, CW, 1,  1.f, 0);
    gemm(gy, 0, CW, 0, Aim_i, 0, nullptr, br,      0, br, 0, ROWS, CW, 1, -1.f, 0);
    gemm(gy, 0, CW, 0, Are_i, 0, nullptr, nullptr, 0, bi, 0, ROWS, CW, 1,  1.f, 0);
    gemm(gx, 0, CW, 0, Aim_i, 0, nullptr, bi,      0, bi, 0, ROWS, CW, 1,  1.f, 0);
    // f = [x | x_diff | tanh(gx*br + gy*bi)]
    gfeat_concat_kernel<<<nel / 256, 256, 0, stream>>>(xb, xdiff, gx, gy, br, bi, fc, nel);
    // MLP + residual
    gemm(fc, 0, 3 * CW, 0, W0 + (size_t)i * 3 * CW * CW, 0, b0 + i * CW,
         nullptr, 0, xm /*h0*/, 0, ROWS, 3 * CW, 1, 1.f, 1);
    gemm(xm, 0, CW, 0, W1 + (size_t)i * CW * CW, 0, b1 + i * CW,
         nullptr, 0, gx /*h1*/, 0, ROWS, CW, 1, 1.f, 1);
    gemm(gx, 0, CW, 0, W2 + (size_t)i * CW * CW, 0, b2 + i * CW,
         xb, 0, xb, 0, ROWS, CW, 1, 1.f, 0);
  }

  // out = x @ Wl + bl
  gemm(xb, 0, CW, 0, Wl, 0, bl, nullptr, 0, (float*)d_out, 0, ROWS, CW, 1, 1.f, 0);
}
